// GCNNetDirected_67336497266904
// MI455X (gfx1250) — matrix operand order repaired and checked by reading, not compiled, not run on hardware
//
#include <hip/hip_runtime.h>
#include <hip/hip_bf16.h>

#define NN 50000
#define NE 800000
#define DIN 128
#define HID2 512
#define DOUT 64

typedef __attribute__((ext_vector_type(16))) __bf16 v16bf;
typedef __attribute__((ext_vector_type(8)))  float  v8f;

union FragU { v16bf v; float4 f[2]; };

// A fragment (16x32 bf16, row-major source):
// lane (hi,m): VGPR0-3 = K[hi*8 .. hi*8+7], VGPR4-7 = K[16+hi*8 .. 16+hi*8+7]
__device__ __forceinline__ v16bf load_a_frag(const __bf16* row, int hi) {
  FragU u;
  u.f[0] = *(const float4*)(row + hi * 8);
  u.f[1] = *(const float4*)(row + 16 + hi * 8);
  return u.v;
}
// B fragment (32x16 bf16) from B^T row (row-major [N,K] layout):
// lane (hi,n): VGPR0-3 = K[hi*8 .. hi*8+7], VGPR4-7 = K[16+hi*8 .. 16+hi*8+7]
__device__ __forceinline__ v16bf load_b_frag_lds(const __bf16* row, int hi) {
  FragU u;
  u.f[0] = *(const float4*)(row + hi * 8);
  u.f[1] = *(const float4*)(row + 16 + hi * 8);
  return u.v;
}

__global__ __launch_bounds__(256)
void degrees_kernel(const int* __restrict__ src, const int* __restrict__ dst,
                    float* __restrict__ degO, float* __restrict__ degI, int nE) {
  int e = blockIdx.x * blockDim.x + threadIdx.x;
  if (e < nE) {
    __hip_atomic_fetch_add(&degO[src[e]], 1.0f, __ATOMIC_RELAXED, __HIP_MEMORY_SCOPE_AGENT);
    __hip_atomic_fetch_add(&degI[dst[e]], 1.0f, __ATOMIC_RELAXED, __HIP_MEMORY_SCOPE_AGENT);
  }
}

// one wave per edge: gather x[src] row, scale, atomic scatter into out[dst]
__global__ __launch_bounds__(256)
void conv_scatter_kernel(const float* __restrict__ x, const int* __restrict__ src,
                         const int* __restrict__ dst, const float* __restrict__ degO,
                         const float* __restrict__ degI, float* __restrict__ out,
                         int ncols, int nE) {
  int e = (blockIdx.x * blockDim.x + threadIdx.x) >> 5;
  if (e >= nE) return;
  int lane = threadIdx.x & 31;
  int s = src[e], d = dst[e];
  float a = degO[s]; a = a < 1.0f ? 1.0f : a;
  float b = degI[d]; b = b < 1.0f ? 1.0f : b;
  float norm = rsqrtf(a * b);
  const float* xr = x + (size_t)s * ncols;
  float* orow = out + (size_t)d * ncols;
  for (int c = lane * 4; c < ncols; c += 128) {
    float4 v = *(const float4*)(xr + c);
    __hip_atomic_fetch_add(orow + c + 0, v.x * norm, __ATOMIC_RELAXED, __HIP_MEMORY_SCOPE_AGENT);
    __hip_atomic_fetch_add(orow + c + 1, v.y * norm, __ATOMIC_RELAXED, __HIP_MEMORY_SCOPE_AGENT);
    __hip_atomic_fetch_add(orow + c + 2, v.z * norm, __ATOMIC_RELAXED, __HIP_MEMORY_SCOPE_AGENT);
    __hip_atomic_fetch_add(orow + c + 3, v.w * norm, __ATOMIC_RELAXED, __HIP_MEMORY_SCOPE_AGENT);
  }
}

__global__ __launch_bounds__(256)
void elu_to_bf16_kernel(const float* __restrict__ in, __bf16* __restrict__ out, int n) {
  int i = blockIdx.x * blockDim.x + threadIdx.x;
  if (i < n) {
    float x = in[i];
    float y = x > 0.0f ? x : (expf(x) - 1.0f);
    out[i] = (__bf16)y;
  }
}

__global__ __launch_bounds__(256)
void f32_to_bf16_kernel(const float* __restrict__ in, __bf16* __restrict__ out, int n) {
  int i = blockIdx.x * blockDim.x + threadIdx.x;
  if (i < n) out[i] = (__bf16)in[i];
}

// W1 [2*DIN, HID2] f32 -> fold halves + transpose -> [HID2, DIN] bf16
__global__ __launch_bounds__(256)
void prep_w1_kernel(const float* __restrict__ W1, __bf16* __restrict__ W1cT) {
  int t = blockIdx.x * blockDim.x + threadIdx.x;
  if (t >= HID2 * DIN) return;
  int n = t >> 7;
  int k = t & 127;
  W1cT[(size_t)n * DIN + k] =
      (__bf16)(W1[(size_t)k * HID2 + n] + W1[(size_t)(k + DIN) * HID2 + n]);
}

// W2 [HID2, DOUT] f32 -> transpose -> [DOUT, HID2] bf16
__global__ __launch_bounds__(256)
void prep_w2_kernel(const float* __restrict__ W2, __bf16* __restrict__ W2T) {
  int t = blockIdx.x * blockDim.x + threadIdx.x;
  if (t >= DOUT * HID2) return;
  int n = t >> 9;
  int k = t & 511;
  W2T[(size_t)n * HID2 + k] = (__bf16)W2[(size_t)k * DOUT + n];
}

// C[M, Ntot] = A[M,K]_bf16 x BT[Ntot,K]_bf16^T + bias, f32 accumulate via WMMA.
// Block = 8 waves. Each wave owns a 32(M) x 64(N) strip (two 16-row tiles), so
// every LDS B fragment feeds two WMMAs. B is staged in LDS per 128-wide K-chunk
// and shared by all 8 waves. B fragments are software-pipelined one step ahead,
// A fragments one full k-step ahead.
template <int K>
__global__ __launch_bounds__(256)
void gemm_bf16_wmma_kernel(const __bf16* __restrict__ A, const __bf16* __restrict__ BT,
                           const float* __restrict__ bias, float* __restrict__ C,
                           int M, int Ntot) {
  constexpr int KC  = 128;       // K-chunk
  constexpr int LDB = KC + 8;    // padded row stride (bf16): 272B = 68 dwords
                                 // -> 16 lanes rotate across all 64 LDS banks
  __shared__ __align__(16) __bf16 Bs[64 * LDB];   // 17408 B

  const int lane = threadIdx.x & 31;
  const int wave = threadIdx.x >> 5;
  const int nbase = blockIdx.y * 64;
  const int pairIdx = blockIdx.x * 8 + wave;
  const int mt0 = pairIdx * 2;
  const int mt1 = mt0 + 1;
  const bool act0 = (mt0 * 16 < M);   // wave-uniform
  const bool act1 = (mt1 * 16 < M);   // act1 implies act0
  const int nl = lane & 15;
  const int hi = lane >> 4;

  const __bf16* aRow0 = A + (size_t)((act0 ? mt0 * 16 : 0) + nl) * K;
  const __bf16* aRow1 = A + (size_t)((act1 ? mt1 * 16 : 0) + nl) * K;

  v8f acc0[4], acc1[4];
  v8f z = {0.f, 0.f, 0.f, 0.f, 0.f, 0.f, 0.f, 0.f};
#pragma unroll
  for (int j = 0; j < 4; ++j) { acc0[j] = z; acc1[j] = z; }

  for (int kc = 0; kc < K; kc += KC) {
    // cooperative stage of B^T chunk [64 x 128] (all 256 threads)
#pragma unroll
    for (int i = 0; i < 4; ++i) {
      int idx = threadIdx.x + i * 256;       // 1024 float4 pieces
      int r = idx >> 4;                      // row 0..63
      int c = (idx & 15) * 8;                // bf16 col 0..120
      *(float4*)(&Bs[r * LDB + c]) =
          *(const float4*)(BT + (size_t)(nbase + r) * K + kc + c);
    }
    __syncthreads();

    if (act0) {
      v16bf a0c = load_a_frag(aRow0 + kc, hi);
      v16bf a1c = load_a_frag(aRow1 + kc, hi);
      v16bf a0n = a0c, a1n = a1c;
      v16bf bc  = load_b_frag_lds(&Bs[nl * LDB], hi);
#pragma unroll
      for (int t = 0; t < 16; ++t) {
        const int kk = t >> 2;
        const int j  = t & 3;
        // prefetch next k-step's A fragments a full k-step early
        if (j == 0 && kk < 3) {
          a0n = load_a_frag(aRow0 + kc + (kk + 1) * 32, hi);
          a1n = load_a_frag(aRow1 + kc + (kk + 1) * 32, hi);
        }
        // prefetch next B fragment one step early
        v16bf bn = bc;
        if (t < 15) {
          const int t1 = t + 1;
          bn = load_b_frag_lds(&Bs[((t1 & 3) * 16 + nl) * LDB + (t1 >> 2) * 32], hi);
        }
        acc0[j] = __builtin_amdgcn_wmma_f32_16x16x32_bf16(
            false, a0c, false, bc, (short)0, acc0[j], false, false);
        acc1[j] = __builtin_amdgcn_wmma_f32_16x16x32_bf16(
            false, a1c, false, bc, (short)0, acc1[j], false, false);
        if (j == 3) { a0c = a0n; a1c = a1n; }
        bc = bn;
      }
    }
    __syncthreads();
  }

  if (act0) {
#pragma unroll
    for (int j = 0; j < 4; ++j) {
      int col = nbase + j * 16 + nl;
      float bv = bias[col];
      float* c0 = C + (size_t)(mt0 * 16 + hi * 8) * Ntot + col;
#pragma unroll
      for (int v = 0; v < 8; ++v) {
        c0[(size_t)v * Ntot] = acc0[j][v] + bv;
      }
      if (act1) {
        float* c1 = C + (size_t)(mt1 * 16 + hi * 8) * Ntot + col;
#pragma unroll
        for (int v = 0; v < 8; ++v) {
          c1[(size_t)v * Ntot] = acc1[j][v] + bv;
        }
      }
    }
  }
}

extern "C" void kernel_launch(void* const* d_in, const int* in_sizes, int n_in,
                              void* d_out, int out_size, void* d_ws, size_t ws_size,
                              hipStream_t stream) {
  const float* feature = (const float*)d_in[0];
  const int*   edge    = (const int*)d_in[1];
  const float* W1      = (const float*)d_in[2];
  const float* b1      = (const float*)d_in[3];
  const float* W2      = (const float*)d_in[4];
  const float* b2      = (const float*)d_in[5];
  const int* src = edge;
  const int* dst = edge + NE;

  char* ws = (char*)d_ws;
  size_t off = 0;
  auto take = [&](size_t bytes) -> void* {
    off = (off + 255) & ~(size_t)255;
    void* p = ws + off;
    off += bytes;
    return p;
  };

  float*  degO  = (float*)take((size_t)NN * 4);
  float*  degI  = (float*)take((size_t)NN * 4);
  float*  agg1  = (float*)take((size_t)NN * DIN * 4);     // conv1 accumulator
  __bf16* hbf   = (__bf16*)take((size_t)NN * DIN * 2);    // ELU(conv1) bf16
  __bf16* W1cT  = (__bf16*)take((size_t)HID2 * DIN * 2);  // folded W1^T bf16
  __bf16* W2T   = (__bf16*)take((size_t)DOUT * HID2 * 2); // W2^T bf16
  float*  H2    = (float*)take((size_t)NN * HID2 * 4);    // linear1 output
  float*  H3    = (float*)take((size_t)NN * HID2 * 4);    // conv2 accumulator
  __bf16* H3bf  = (__bf16*)take((size_t)NN * HID2 * 2);   // conv2 output bf16

  hipMemsetAsync(degO, 0, (size_t)NN * 4, stream);
  hipMemsetAsync(degI, 0, (size_t)NN * 4, stream);
  hipMemsetAsync(agg1, 0, (size_t)NN * DIN * 4, stream);
  hipMemsetAsync(H3,   0, (size_t)NN * HID2 * 4, stream);

  degrees_kernel<<<(NE + 255) / 256, 256, 0, stream>>>(src, dst, degO, degI, NE);

  conv_scatter_kernel<<<(NE * 32 + 255) / 256, 256, 0, stream>>>(
      feature, src, dst, degO, degI, agg1, DIN, NE);

  elu_to_bf16_kernel<<<(NN * DIN + 255) / 256, 256, 0, stream>>>(agg1, hbf, NN * DIN);

  prep_w1_kernel<<<(HID2 * DIN + 255) / 256, 256, 0, stream>>>(W1, W1cT);
  prep_w2_kernel<<<(DOUT * HID2 + 255) / 256, 256, 0, stream>>>(W2, W2T);

  {
    int pairTiles = (NN / 16 + 1) / 2;             // 1563 pairs
    dim3 grid((pairTiles + 7) / 8, HID2 / 64);     // (196, 8)
    gemm_bf16_wmma_kernel<DIN><<<grid, 256, 0, stream>>>(hbf, W1cT, b1, H2, NN, HID2);
  }

  conv_scatter_kernel<<<(NE * 32 + 255) / 256, 256, 0, stream>>>(
      H2, src, dst, degO, degI, H3, HID2, NE);

  f32_to_bf16_kernel<<<(NN * HID2 + 255) / 256, 256, 0, stream>>>(H3, H3bf, NN * HID2);

  {
    int pairTiles = (NN / 16 + 1) / 2;
    dim3 grid((pairTiles + 7) / 8, DOUT / 64);     // (196, 1)
    gemm_bf16_wmma_kernel<HID2><<<grid, 256, 0, stream>>>(
        H3bf, W2T, b2, (float*)d_out, NN, DOUT);
  }
}
